// GCN_two_89704686944356
// MI455X (gfx1250) — matrix operand order repaired and checked by reading, not compiled, not run on hardware
//
#include <hip/hip_runtime.h>
#include <math.h>

#define N_NODES 100000
#define N_FEAT  500
#define N_HID   16
#define N_CLS   7
#define N_EDGES 3200000
#define N_TILES (N_NODES / 16)   // 6250 exactly

typedef __attribute__((ext_vector_type(16))) _Float16 v16h;
typedef __attribute__((ext_vector_type(8)))  float    v8f;

__device__ __forceinline__ void atomic_add_f32(float* p, float v) {
    __hip_atomic_fetch_add(p, v, __ATOMIC_RELAXED, __HIP_MEMORY_SCOPE_AGENT);
}

// ---------------------------------------------------------------------------
// init: deg = 1.0 (self-loop), agg1 = 0, out = 0   (1.6M threads covers all)
// ---------------------------------------------------------------------------
__global__ void init_kernel(float* __restrict__ deg, float* __restrict__ agg1,
                            float* __restrict__ out) {
    int idx = blockIdx.x * blockDim.x + threadIdx.x;
    if (idx < N_NODES)          deg[idx]  = 1.0f;
    if (idx < N_NODES * N_HID)  agg1[idx] = 0.0f;
    if (idx < N_NODES * N_CLS)  out[idx]  = 0.0f;
}

// ---------------------------------------------------------------------------
// in-degree accumulation over dst
// ---------------------------------------------------------------------------
__global__ void degree_kernel(const int* __restrict__ dst, float* __restrict__ deg) {
    int e = blockIdx.x * blockDim.x + threadIdx.x;
    if (e >= N_EDGES) return;
    atomic_add_f32(&deg[dst[e]], 1.0f);
}

__global__ void rsqrt_kernel(float* __restrict__ dinv) {
    int i = blockIdx.x * blockDim.x + threadIdx.x;
    if (i >= N_NODES) return;
    dinv[i] = rsqrtf(dinv[i]);   // deg >= 1 always
}

// ---------------------------------------------------------------------------
// GEMM1: m1[100000x16] = x[100000x500] @ W1[500x16]
// One wave per 16-row tile. W1 is staged once per block into LDS in the
// WMMA B-fragment layout (f16, zero-padded past K=500). Branch-free K-tail.
// ---------------------------------------------------------------------------
__global__ __launch_bounds__(256) void gemm1_wmma(const float* __restrict__ x,
                                                  const float* __restrict__ W1,
                                                  float* __restrict__ m1) {
    // B fragments in LDS: [kc][lane][elem] -> 16*32*16 halves = 16 KB
    __shared__ __align__(32) _Float16 Bsh[16 * 32 * 16];

    // cooperative fill: elem i of lane l (g=l>>4,lm=l&15), chunk kc holds
// f16(W1[k*16+lm]) with k = kc*32 + (i<8 ? g*8+i : 16+g*8+i-8), zero for k >= 500.
    for (int n = threadIdx.x; n < 16 * 32 * 16; n += 256) {
        const int kc = n >> 9;
        const int l  = (n >> 4) & 31;
        const int i  = n & 15;
        const int gg = l >> 4, ll = l & 15;
        const int k  = kc * 32 + (i < 8 ? gg * 8 + i : 16 + gg * 8 + (i - 8));
        const float w = W1[(k < N_FEAT ? k : 0) * N_HID + ll];
        Bsh[n] = (_Float16)(k < N_FEAT ? w : 0.f);
    }
    __syncthreads();

    const int lane = threadIdx.x & 31;
    const int tile = blockIdx.x * 8 + (threadIdx.x >> 5);
    if (tile >= N_TILES) return;                 // wave-uniform exit
    const int g  = lane >> 4;                    // lane half-group
    const int lm = lane & 15;
    const float* xrow = x + (size_t)(tile * 16 + lm) * N_FEAT;
    const _Float16* bbase = &Bsh[lane << 4];

    v8f c = {0.f, 0.f, 0.f, 0.f, 0.f, 0.f, 0.f, 0.f};

    // 15 full K-chunks of 32 (k up to 479 < 500, no guard needed)
    for (int kc = 0; kc < 15; ++kc) {
        const int k0 = kc * 32;
        const int ka = k0 + g * 8;               // A: elems 0..7 -> ka.., 8..15 -> ka+16..
        float4 p0 = *(const float4*)(xrow + ka);
        float4 p1 = *(const float4*)(xrow + ka + 4);
        float4 p2 = *(const float4*)(xrow + ka + 16);
        float4 p3 = *(const float4*)(xrow + ka + 20);
        v16h a;
        a[0]=(_Float16)p0.x;  a[1]=(_Float16)p0.y;  a[2]=(_Float16)p0.z;  a[3]=(_Float16)p0.w;
        a[4]=(_Float16)p1.x;  a[5]=(_Float16)p1.y;  a[6]=(_Float16)p1.z;  a[7]=(_Float16)p1.w;
        a[8]=(_Float16)p2.x;  a[9]=(_Float16)p2.y;  a[10]=(_Float16)p2.z; a[11]=(_Float16)p2.w;
        a[12]=(_Float16)p3.x; a[13]=(_Float16)p3.y; a[14]=(_Float16)p3.z; a[15]=(_Float16)p3.w;
        const v16h b = *(const v16h*)(bbase + (kc << 9));
        c = __builtin_amdgcn_wmma_f32_16x16x32_f16(false, a, false, b, (short)0, c,
                                                   false, false);
    }

    // tail chunk kc=15 (k0=480): elems 0..7 statically in-bounds (max k=495);
    // elems 8..15 use clamped address + value select (branch-free).
    {
        const int ka = 480 + g * 8;
        float4 q0 = *(const float4*)(xrow + ka);
        float4 q1 = *(const float4*)(xrow + ka + 4);
        v16h a;
        a[0]=(_Float16)q0.x; a[1]=(_Float16)q0.y; a[2]=(_Float16)q0.z; a[3]=(_Float16)q0.w;
        a[4]=(_Float16)q1.x; a[5]=(_Float16)q1.y; a[6]=(_Float16)q1.z; a[7]=(_Float16)q1.w;
        #pragma unroll
        for (int i = 0; i < 8; ++i) {
            const int k2 = 496 + g * 8 + i;               // 496..511
            const float v = xrow[k2 < N_FEAT ? k2 : (N_FEAT - 1)];
            a[i + 8] = (_Float16)(k2 < N_FEAT ? v : 0.f);
        }
        const v16h b = *(const v16h*)(bbase + (15 << 9));
        c = __builtin_amdgcn_wmma_f32_16x16x32_f16(false, a, false, b, (short)0, c,
                                                   false, false);
    }

    // D layout: VGPR r -> row (g*8 + r), col lm
    #pragma unroll
    for (int r = 0; r < 8; ++r)
        m1[(size_t)(tile * 16 + g * 8 + r) * N_HID + lm] = c[r];
}

// ---------------------------------------------------------------------------
// scatter 1: agg1[dst] += m1[src] * dinv[src]*dinv[dst]; 16 lanes per edge
// ---------------------------------------------------------------------------
__global__ void agg1_kernel(const int* __restrict__ src, const int* __restrict__ dst,
                            const float* __restrict__ dinv,
                            const float* __restrict__ m1, float* __restrict__ agg1) {
    int idx = blockIdx.x * blockDim.x + threadIdx.x;   // E*16 = 51.2M threads
    int e = idx >> 4, j = idx & 15;
    int s = src[e], d = dst[e];
    float w = dinv[s] * dinv[d];
    atomic_add_f32(&agg1[(size_t)d * N_HID + j], m1[(size_t)s * N_HID + j] * w);
}

// ---------------------------------------------------------------------------
// self-loop + bias + ReLU:  h = relu(agg1 + m1*dinv^2 + b1)   (in place)
// ---------------------------------------------------------------------------
__global__ void bias_relu_kernel(float* __restrict__ h, const float* __restrict__ m1,
                                 const float* __restrict__ dinv,
                                 const float* __restrict__ b1) {
    int idx = blockIdx.x * blockDim.x + threadIdx.x;
    if (idx >= N_NODES * N_HID) return;
    int v = idx >> 4, j = idx & 15;
    float dv = dinv[v];
    float val = h[idx] + m1[idx] * dv * dv + b1[j];
    h[idx] = val > 0.f ? val : 0.f;
}

// ---------------------------------------------------------------------------
// GEMM2: m2[100000x7] = h[100000x16] @ W2[16x7]; K padded 16->32, N 7->16
// ---------------------------------------------------------------------------
__global__ __launch_bounds__(256) void gemm2_wmma(const float* __restrict__ h,
                                                  const float* __restrict__ W2,
                                                  float* __restrict__ m2) {
    const int lane = threadIdx.x & 31;
    const int tile = blockIdx.x * 8 + (threadIdx.x >> 5);
    if (tile >= N_TILES) return;
    const int g  = lane >> 4;
    const int lm = lane & 15;
    const float* hrow = h + (size_t)(tile * 16 + lm) * N_HID;

    float4 h0 = *(const float4*)(hrow + g * 8);
    float4 h1 = *(const float4*)(hrow + g * 8 + 4);
    v16h a, b;
    a[0]=(_Float16)h0.x; a[1]=(_Float16)h0.y; a[2]=(_Float16)h0.z; a[3]=(_Float16)h0.w;
    a[4]=(_Float16)h1.x; a[5]=(_Float16)h1.y; a[6]=(_Float16)h1.z; a[7]=(_Float16)h1.w;
    #pragma unroll
    for (int i = 0; i < 8; ++i) a[i + 8] = (_Float16)0.f;   // K = 16..31 zero-pad
    #pragma unroll
    for (int i = 0; i < 16; ++i) {                // i<8: K=g*8+i; i>=8: K>=16 -> 0
        float w = (i < 8 && lm < N_CLS) ? W2[(g * 8 + i) * N_CLS + lm] : 0.f;
        b[i] = (_Float16)w;
    }
    v8f c = {0.f, 0.f, 0.f, 0.f, 0.f, 0.f, 0.f, 0.f};
    c = __builtin_amdgcn_wmma_f32_16x16x32_f16(false, a, false, b, (short)0, c,
                                               false, false);
    if (lm < N_CLS) {
        #pragma unroll
        for (int r = 0; r < 8; ++r)
            m2[(size_t)(tile * 16 + g * 8 + r) * N_CLS + lm] = c[r];
    }
}

// ---------------------------------------------------------------------------
// scatter 2: out[dst] += m2[src] * norm; 8 lanes per edge (7 active)
// ---------------------------------------------------------------------------
__global__ void agg2_kernel(const int* __restrict__ src, const int* __restrict__ dst,
                            const float* __restrict__ dinv,
                            const float* __restrict__ m2, float* __restrict__ out) {
    int idx = blockIdx.x * blockDim.x + threadIdx.x;   // E*8 = 25.6M threads
    int e = idx >> 3, j = idx & 7;
    if (j >= N_CLS) return;
    int s = src[e], d = dst[e];
    float w = dinv[s] * dinv[d];
    atomic_add_f32(&out[(size_t)d * N_CLS + j], m2[(size_t)s * N_CLS + j] * w);
}

// ---------------------------------------------------------------------------
// self-loop + bias + log_softmax over 7 classes (in place on out)
// ---------------------------------------------------------------------------
__global__ void logsoftmax_kernel(float* __restrict__ out, const float* __restrict__ m2,
                                  const float* __restrict__ dinv,
                                  const float* __restrict__ b2) {
    int v = blockIdx.x * blockDim.x + threadIdx.x;
    if (v >= N_NODES) return;
    float dv = dinv[v];
    float d2 = dv * dv;
    float z[N_CLS];
    float mx = -INFINITY;
    #pragma unroll
    for (int j = 0; j < N_CLS; ++j) {
        z[j] = out[(size_t)v * N_CLS + j] + m2[(size_t)v * N_CLS + j] * d2 + b2[j];
        mx = fmaxf(mx, z[j]);
    }
    float s = 0.f;
    #pragma unroll
    for (int j = 0; j < N_CLS; ++j) s += __expf(z[j] - mx);
    float lse = mx + __logf(s);
    #pragma unroll
    for (int j = 0; j < N_CLS; ++j) out[(size_t)v * N_CLS + j] = z[j] - lse;
}

// ---------------------------------------------------------------------------
extern "C" void kernel_launch(void* const* d_in, const int* in_sizes, int n_in,
                              void* d_out, int out_size, void* d_ws, size_t ws_size,
                              hipStream_t stream) {
    const float* x  = (const float*)d_in[0];
    const int*   ei = (const int*)d_in[1];
    const float* W1 = (const float*)d_in[2];
    const float* b1 = (const float*)d_in[3];
    const float* W2 = (const float*)d_in[4];
    const float* b2 = (const float*)d_in[5];
    float* out = (float*)d_out;

    const int* src = ei;             // edge_index[0]
    const int* dst = ei + N_EDGES;   // edge_index[1]

    // workspace layout (floats): dinv/deg | m1 | agg1->h | m2   (16 MB total)
    float* ws   = (float*)d_ws;
    float* dinv = ws;                                   // 100000
    float* m1   = ws + 100000;                          // 1,600,000
    float* h    = ws + 1700000;                         // 1,600,000 (agg1, then h)
    float* m2   = ws + 3300000;                         // 700,000

    init_kernel<<<(N_NODES * N_HID + 255) / 256, 256, 0, stream>>>(dinv, h, out);
    degree_kernel<<<(N_EDGES + 255) / 256, 256, 0, stream>>>(dst, dinv);
    rsqrt_kernel<<<(N_NODES + 255) / 256, 256, 0, stream>>>(dinv);

    gemm1_wmma<<<(N_TILES + 7) / 8, 256, 0, stream>>>(x, W1, m1);
    agg1_kernel<<<(N_EDGES * 16) / 256, 256, 0, stream>>>(src, dst, dinv, m1, h);
    bias_relu_kernel<<<(N_NODES * N_HID + 255) / 256, 256, 0, stream>>>(h, m1, dinv, b1);

    gemm2_wmma<<<(N_TILES + 7) / 8, 256, 0, stream>>>(h, W2, m2);
    agg2_kernel<<<(N_EDGES * 8) / 256, 256, 0, stream>>>(src, dst, dinv, m2, out);
    logsoftmax_kernel<<<(N_NODES + 255) / 256, 256, 0, stream>>>(out, m2, dinv, b2);
}
